// FourierStage_51230369906844
// MI455X (gfx1250) — matrix operand order repaired and checked by reading, not compiled, not run on hardware
//
#include <hip/hip_runtime.h>
#include <hip/hip_bf16.h>

typedef __attribute__((ext_vector_type(16))) _Float16 v16h;
typedef __attribute__((ext_vector_type(8)))  float    v8f;

#define B_   2
#define H_   128
#define W_   128
#define NPIX (B_ * H_ * W_)   // 32768

// ---------------------------------------------------------------------------
// WMMA helpers (gfx1250, wave32)
// ---------------------------------------------------------------------------
__device__ inline v8f wmma32(v16h a, v16h b, v8f c) {
  // v_wmma_f32_16x16x32_f16 : D = A(16x32) * B(32x16) + C
  return __builtin_amdgcn_wmma_f32_16x16x32_f16(
      false, a, false, b, (short)0, c, false, false);
}

// A fragment from a row-major f16 LDS matrix [rows][ldr].
// ISA 16-bit A layout: lane m = lane&15 is the row; lanes<16 hold K {0..7,16..23},
// lanes>=16 hold K {8..15,24..31} (relative to k_off).
__device__ inline v16h ldsA(const _Float16* base, int ldr, int m_off, int k_off,
                            int lane) {
  int m = m_off + (lane & 15);
  const _Float16* row = base + m * ldr + k_off + ((lane >> 4) * 8);
  v16h a;
#pragma unroll
  for (int v = 0; v < 8; ++v) {
    int k = ((v < 4) ? 0 : 16) + (v & 3) * 2;
    a[2 * v]     = row[k];
    a[2 * v + 1] = row[k + 1];
  }
  return a;
}

// B fragment from a K-transposed f16 LDS matrix Bt[n][ldk] (k contiguous).
// ISA 16-bit B layout: lane n = lane&15 is the column; lanes<16 hold K {0..7,16..23},
// lanes>=16 hold K {8..15,24..31} (relative to k_off).
__device__ inline v16h ldsB(const _Float16* baseT, int ldk, int n_off, int k_off,
                            int lane) {
  int n = n_off + (lane & 15);
  const _Float16* col = baseT + n * ldk + k_off + ((lane >> 4) * 8);
  v16h b;
#pragma unroll
  for (int v = 0; v < 8; ++v) {
    int k = ((v < 4) ? 0 : 16) + (v & 3) * 2;
    b[2 * v]     = col[k];
    b[2 * v + 1] = col[k + 1];
  }
  return b;
}

// A fragment for the fused projection GEMM: z[p, k*32+c] = h2[p,k]*x[p,c],
// rows 1024..1055 are the bias rows fed by x directly.
__device__ inline v16h makeAz(const float* h2w, const float* xw, int k_off,
                              int lane) {
  int m = lane & 31 & 15, half = lane >> 4;
  m = lane & 15;
  v16h a;
#pragma unroll
  for (int v = 0; v < 8; ++v) {
#pragma unroll
    for (int e = 0; e < 2; ++e) {
      int kk = k_off + ((v < 4) ? 0 : 16) + half * 8 + (v & 3) * 2 + e;
      float z;
      if (kk < 1024) z = h2w[m * 32 + (kk >> 5)] * xw[m * 32 + (kk & 31)];
      else           z = xw[m * 32 + (kk - 1024)];
      a[2 * v + e] = (_Float16)z;
    }
  }
  return a;
}

__device__ inline float gelu_tanh(float v) {
  float t = 0.7978845608028654f * (v + 0.044715f * v * v * v);
  return 0.5f * v * (1.0f + tanhf(t));
}

// Async global->LDS copy of one 16-byte chunk per lane (CDNA5 path, ASYNCcnt).
__device__ inline void async_copy_b128(const _Float16* gsrc, _Float16* ldst) {
  unsigned lo = (unsigned)(unsigned long long)ldst;   // LDS aperture: low 32 bits
  asm volatile("global_load_async_to_lds_b128 %0, %1, off"
               :: "v"(lo), "v"(gsrc) : "memory");
}
__device__ inline void async_wait0() {
  asm volatile("s_wait_asynccnt 0x0" ::: "memory");
}

struct PP { const float *w1, *b1, *w2, *b2, *w3, *b3; };

// ---------------------------------------------------------------------------
// Twiddle tables, f16, written once:
//   [0]      Fr = cos(2*pi*k*j/128)
//   [16384]  Fi = -sin(...)   (imag part of forward F)
//   [32768]  Fm = +sin(...)   (= -Fi)
// ---------------------------------------------------------------------------
__global__ __launch_bounds__(256) void twiddle_k(_Float16* __restrict__ Ftab) {
  int i = blockIdx.x * 256 + threadIdx.x;
  if (i < 16384) {
    int k = i >> 7, j = i & 127;
    int t = (k * j) & 127;                      // exact periodic reduction
    float th = (float)t * (6.283185307179586f / 128.0f);
    float c = cosf(th), s = sinf(th);
    Ftab[i]         = (_Float16)c;
    Ftab[16384 + i] = (_Float16)(-s);
    Ftab[32768 + i] = (_Float16)s;
  }
}

// ---------------------------------------------------------------------------
// Stage A: fused projections g2,g4,g3 + per-pixel D() application.
//   u[p, 0:32]  = D(x, gamma2(sos))
//   u[p, 32:64] = D(x, gamma4(den))
//   xl[p, 0:32] = D(x, gamma3(cat(sos,den)))
// ---------------------------------------------------------------------------
__global__ __launch_bounds__(256) void stage_a(
    const float* __restrict__ x, const float* __restrict__ sos,
    const float* __restrict__ den, PP p2, PP p4, PP p3,
    float* __restrict__ u, float* __restrict__ xl) {
  __shared__ _Float16 Wt[32 * 1056];                      // 66 KB (transposed W3e)
  __shared__ float w1s[32 * 32], w2s[32 * 32], b1s[32], b2s[32];
  __shared__ float ctx0[8][256], ctx1[8][256];
  __shared__ float xin[8][512], h1[8][512], h2[8][512];

  int tid = threadIdx.x, lane = tid & 31, wv = tid >> 5;
  int tp = blockIdx.x * 128 + wv * 16;                    // tile pixel base

  for (int i = lane; i < 256; i += 32) {
    int p = i >> 4, j = i & 15;
    ctx0[wv][i] = sos[(tp + p) * 16 + j];
    ctx1[wv][i] = den[(tp + p) * 16 + j];
  }
  for (int i = lane; i < 512; i += 32) {
    int p = i >> 5, c = i & 31;
    xin[wv][i] = x[(tp + p) * 32 + c];
  }

  PP prj[3] = {p2, p4, p3};
  const int cins[3] = {16, 16, 32};

  for (int pr = 0; pr < 3; ++pr) {
    __syncthreads();                                      // Wt safe to overwrite
    PP P = prj[pr];
    int cin = cins[pr];
    __builtin_prefetch(P.w3 + tid * 32, 0, 1);            // global_prefetch_b8
    for (int i = tid; i < cin * 32; i += 256) w1s[i] = P.w1[i];
    for (int i = tid; i < 1024; i += 256)     w2s[i] = P.w2[i];
    if (tid < 32) { b1s[tid] = P.b1[tid]; b2s[tid] = P.b2[tid]; }
    for (int i = tid; i < 32 * 1056; i += 256) {
      int o = i / 1056, kk = i - o * 1056;
      float v = (kk < 1024) ? P.w3[(kk >> 5) * 1024 + (kk & 31) * 32 + o]
                            : P.b3[(kk - 1024) * 32 + o];
      Wt[o * 1056 + kk] = (_Float16)v;
    }
    __syncthreads();

    // L1: h1 = exp(-(ctx @ w1 + b1)^2), lane = output channel
    {
      int o = lane;
      for (int p = 0; p < 16; ++p) {
        float s = b1s[o];
        for (int j = 0; j < cin; ++j) {
          float in;
          if (pr == 2) in = (j < 16) ? ctx0[wv][p * 16 + j]
                                     : ctx1[wv][p * 16 + j - 16];
          else in = (pr == 0) ? ctx0[wv][p * 16 + j] : ctx1[wv][p * 16 + j];
          s += in * w1s[j * 32 + o];
        }
        h1[wv][p * 32 + o] = expf(-s * s);
      }
    }
    __syncthreads();
    // L2: h2 = exp(-(h1 @ w2 + b2)^2)
    {
      int o = lane;
      for (int p = 0; p < 16; ++p) {
        float s = b2s[o];
        for (int k = 0; k < 32; ++k) s += h1[wv][p * 32 + k] * w2s[k * 32 + o];
        h2[wv][p * 32 + o] = expf(-s * s);
      }
    }
    __syncthreads();

    // Fused L3 + D():  [16,1056] @ [1056,32]
    v8f acc0 = {}; v8f acc1 = {};
    for (int t = 0; t < 33; ++t) {
      v16h a = makeAz(h2[wv], xin[wv], t * 32, lane);
      acc0 = wmma32(a, ldsB(Wt, 1056, 0,  t * 32, lane), acc0);
      acc1 = wmma32(a, ldsB(Wt, 1056, 16, t * 32, lane), acc1);
    }
    int half = lane >> 4, n = lane & 15;
#pragma unroll
    for (int r = 0; r < 8; ++r) {
      int m = r + 8 * half;
      int pix = tp + m;
      if (pr == 0)      { u[pix * 64 + n]      = acc0[r]; u[pix * 64 + 16 + n] = acc1[r]; }
      else if (pr == 1) { u[pix * 64 + 32 + n] = acc0[r]; u[pix * 64 + 48 + n] = acc1[r]; }
      else              { xl[pix * 32 + n]     = acc0[r]; xl[pix * 32 + 16 + n] = acc1[r]; }
    }
  }
}

// ---------------------------------------------------------------------------
// DFT pass as GEMM.  One block = one 128-length slice (row or column).
// mode: 0 real-in fwd, 1 complex fwd, 2 complex inverse (conj F),
//       3 complex-in real-out inverse (with scale).
// ---------------------------------------------------------------------------
__global__ __launch_bounds__(256) void dft_pass(
    const _Float16* __restrict__ Ftab, const float* __restrict__ in_r,
    const float* __restrict__ in_i, float* __restrict__ out_r,
    float* __restrict__ out_i, int cch, int mode, int colmode, float scale) {
  __shared__ __align__(16) _Float16 Fr[16384], Fi[16384], Fm[16384]; // 96 KB
  __shared__ __align__(16) _Float16 Xr[64 * 128], Xi[64 * 128];      // 32 KB

  int tid = threadIdx.x, lane = tid & 31, wv = tid >> 5;
  int s = blockIdx.x;
  long base; long estride;
  if (!colmode) { base = (long)s * 128 * cch; estride = cch; }
  else {
    int b = s >> 7, k2 = s & 127;
    base = ((long)b * 16384 + k2) * cch;
    estride = (long)128 * cch;
  }

  // Async DMA the f16 twiddle tables straight into LDS (no VGPR round-trip).
  for (int i = tid * 8; i < 16384; i += 256 * 8) {
    async_copy_b128(Ftab + i,         &Fr[i]);
    async_copy_b128(Ftab + 16384 + i, &Fi[i]);
    async_copy_b128(Ftab + 32768 + i, &Fm[i]);
  }
  bool cplx_in = (mode != 0);
  for (int i = tid; i < cch * 128; i += 256) {
    int j = i / cch, c = i - j * cch;
    long a = base + (long)j * estride + c;
    Xr[c * 128 + j] = (_Float16)in_r[a];
    _Float16 xiv = (_Float16)0.0f;
    if (cplx_in) xiv = (_Float16)in_i[a];
    Xi[c * 128 + j] = xiv;
  }
  async_wait0();
  __syncthreads();

  int ntiles = cch >> 4;
  int njobs = 8 * ntiles;
  for (int job = wv; job < njobs; job += 8) {
    int mt = job / ntiles, nt = job - mt * ntiles;
    v8f aR = {}; v8f aI = {};
#pragma unroll
    for (int t = 0; t < 4; ++t) {
      int ko = t * 32;
      v16h fr = ldsA(Fr, 128, mt * 16, ko, lane);
      v16h fi = ldsA(Fi, 128, mt * 16, ko, lane);
      v16h fm = ldsA(Fm, 128, mt * 16, ko, lane);
      v16h xr = ldsB(Xr, 128, nt * 16, ko, lane);
      v16h xi = ldsB(Xi, 128, nt * 16, ko, lane);
      if (mode == 0) {                       // Y = F * x (x real)
        aR = wmma32(fr, xr, aR);
        aI = wmma32(fi, xr, aI);
      } else if (mode == 1) {                // Z = F * X
        aR = wmma32(fr, xr, aR); aR = wmma32(fm, xi, aR);
        aI = wmma32(fr, xi, aI); aI = wmma32(fi, xr, aI);
      } else if (mode == 2) {                // O = conj(F) * X
        aR = wmma32(fr, xr, aR); aR = wmma32(fi, xi, aR);
        aI = wmma32(fr, xi, aI); aI = wmma32(fm, xr, aI);
      } else {                               // real(conj(F) * X) * scale
        aR = wmma32(fr, xr, aR); aR = wmma32(fi, xi, aR);
      }
    }
    int half = lane >> 4, n = nt * 16 + (lane & 15);
#pragma unroll
    for (int r = 0; r < 8; ++r) {
      int m = mt * 16 + r + 8 * half;
      long a = base + (long)m * estride + n;
      out_r[a] = aR[r] * scale;
      if (mode != 3) out_i[a] = aI[r];
    }
  }
}

// ---------------------------------------------------------------------------
// Per-frequency Green's-function apply: uf[b,f,o] = sum_c Z[b,f,c] * g[c,o,f]
// g = (dr + i) / (dr^2 + 1),  dr = softplus(amp)*fsq - softplus(k0)
// ---------------------------------------------------------------------------
__global__ __launch_bounds__(256) void freq_apply(
    const float* __restrict__ Zr, const float* __restrict__ Zi,
    float* __restrict__ Ur, float* __restrict__ Ui,
    const float* __restrict__ k0, const float* __restrict__ amp, int cin) {
  __shared__ float k0s[64 * 32], amps[64 * 32];
  int tid = threadIdx.x;
  for (int i = tid; i < cin * 32; i += 256) {
    k0s[i]  = log1pf(expf(k0[i]));
    amps[i] = log1pf(expf(amp[i]));
  }
  __syncthreads();
  int lane = tid & 31, wv = tid >> 5;
  int o = lane;
  int gw = blockIdx.x * 8 + wv;
  int nw = gridDim.x * 8;
  for (int f = gw; f < 16384; f += nw) {
    int k1 = f >> 7, k2 = f & 127;
    float fx = 6.283185307179586f * ((float)(k1 < 64 ? k1 : k1 - 128) / 128.0f);
    float fy = 6.283185307179586f * ((float)(k2 < 64 ? k2 : k2 - 128) / 128.0f);
    float fsq = fx * fx + fy * fy;
    float ar0 = 0.f, ai0 = 0.f, ar1 = 0.f, ai1 = 0.f;
    for (int c = 0; c < cin; ++c) {
      float dr  = amps[c * 32 + o] * fsq - k0s[c * 32 + o];
      float inv = 1.0f / (dr * dr + 1.0f);
      float gr = dr * inv, gi = inv;
      long i0 = (long)f * cin + c;
      long i1 = (long)(16384 + f) * cin + c;
      float zr = Zr[i0], zi = Zi[i0];
      ar0 += zr * gr - zi * gi; ai0 += zr * gi + zi * gr;
      zr = Zr[i1]; zi = Zi[i1];
      ar1 += zr * gr - zi * gi; ai1 += zr * gi + zi * gr;
    }
    long o0 = (long)f * 32 + o;
    long o1 = (long)(16384 + f) * 32 + o;
    Ur[o0] = ar0; Ui[o0] = ai0;
    Ur[o1] = ar1; Ui[o1] = ai1;
  }
}

// ---------------------------------------------------------------------------
// Stage D: precond = D(c, gamma1(sos)); xf = D(precond, gamma5(den));
// out = dense2(gelu(dense1(xf + x_local)))
// ---------------------------------------------------------------------------
__global__ __launch_bounds__(256) void stage_d(
    const float* __restrict__ sos, const float* __restrict__ den,
    const float* __restrict__ cbuf, const float* __restrict__ xl,
    PP p1, PP p5,
    const float* __restrict__ d1w, const float* __restrict__ d1b,
    const float* __restrict__ d2w, const float* __restrict__ d2b,
    float* __restrict__ out) {
  __shared__ _Float16 Wt[32 * 1056];
  __shared__ float w1s[16 * 32], w2s[32 * 32], b1s[32], b2s[32];
  __shared__ float dw1[32 * 32], dw2[32 * 32], db1[32], db2[32];
  __shared__ float ctx[8][256], xin[8][512], h1[8][512], h2[8][512];

  int tid = threadIdx.x, lane = tid & 31, wv = tid >> 5;
  int tp = blockIdx.x * 128 + wv * 16;

  for (int i = tid; i < 1024; i += 256) { dw1[i] = d1w[i]; dw2[i] = d2w[i]; }
  if (tid < 32) { db1[tid] = d1b[tid]; db2[tid] = d2b[tid]; }
  for (int i = lane; i < 512; i += 32) {
    int p = i >> 5, c = i & 31;
    xin[wv][i] = cbuf[(tp + p) * 32 + c];        // D input for gamma1
  }

  PP prj[2] = {p1, p5};
  for (int pr = 0; pr < 2; ++pr) {
    __syncthreads();
    PP P = prj[pr];
    const float* cx = (pr == 0) ? sos : den;
    __builtin_prefetch(P.w3 + tid * 32, 0, 1);
    for (int i = lane; i < 256; i += 32) {
      int p = i >> 4, j = i & 15;
      ctx[wv][i] = cx[(tp + p) * 16 + j];
    }
    for (int i = tid; i < 512;  i += 256) w1s[i] = P.w1[i];
    for (int i = tid; i < 1024; i += 256) w2s[i] = P.w2[i];
    if (tid < 32) { b1s[tid] = P.b1[tid]; b2s[tid] = P.b2[tid]; }
    for (int i = tid; i < 32 * 1056; i += 256) {
      int o = i / 1056, kk = i - o * 1056;
      float v = (kk < 1024) ? P.w3[(kk >> 5) * 1024 + (kk & 31) * 32 + o]
                            : P.b3[(kk - 1024) * 32 + o];
      Wt[o * 1056 + kk] = (_Float16)v;
    }
    __syncthreads();

    {
      int o = lane;
      for (int p = 0; p < 16; ++p) {
        float s = b1s[o];
        for (int j = 0; j < 16; ++j) s += ctx[wv][p * 16 + j] * w1s[j * 32 + o];
        h1[wv][p * 32 + o] = expf(-s * s);
      }
    }
    __syncthreads();
    {
      int o = lane;
      for (int p = 0; p < 16; ++p) {
        float s = b2s[o];
        for (int k = 0; k < 32; ++k) s += h1[wv][p * 32 + k] * w2s[k * 32 + o];
        h2[wv][p * 32 + o] = expf(-s * s);
      }
    }
    __syncthreads();

    v8f acc0 = {}; v8f acc1 = {};
    for (int t = 0; t < 33; ++t) {
      v16h a = makeAz(h2[wv], xin[wv], t * 32, lane);
      acc0 = wmma32(a, ldsB(Wt, 1056, 0,  t * 32, lane), acc0);
      acc1 = wmma32(a, ldsB(Wt, 1056, 16, t * 32, lane), acc1);
    }
    int half = lane >> 4, n = lane & 15;
#pragma unroll
    for (int r = 0; r < 8; ++r) {
      int m = r + 8 * half;
      if (pr == 0) {                                  // precond -> next D input
        xin[wv][m * 32 + n]      = acc0[r];
        xin[wv][m * 32 + 16 + n] = acc1[r];
      } else {                                        // x_fourier + x_local
        int pix = tp + m;
        h1[wv][m * 32 + n]      = acc0[r] + xl[pix * 32 + n];
        h1[wv][m * 32 + 16 + n] = acc1[r] + xl[pix * 32 + 16 + n];
      }
    }
  }
  __syncthreads();

  {   // dense1 + gelu
    int o = lane;
    for (int p = 0; p < 16; ++p) {
      float s = db1[o];
      for (int c = 0; c < 32; ++c) s += h1[wv][p * 32 + c] * dw1[c * 32 + o];
      h2[wv][p * 32 + o] = gelu_tanh(s);
    }
  }
  __syncthreads();
  {   // dense2
    int o = lane;
    for (int p = 0; p < 16; ++p) {
      float s = db2[o];
      for (int c = 0; c < 32; ++c) s += h2[wv][p * 32 + c] * dw2[c * 32 + o];
      out[(tp + p) * 32 + o] = s;
    }
  }
}

// ---------------------------------------------------------------------------
extern "C" void kernel_launch(void* const* d_in, const int* in_sizes, int n_in,
                              void* d_out, int out_size, void* d_ws,
                              size_t ws_size, hipStream_t stream) {
  const float* x   = (const float*)d_in[0];
  const float* sos = (const float*)d_in[1];
  const float* den = (const float*)d_in[2];
  auto getPP = [&](int i) {
    PP p;
    p.w1 = (const float*)d_in[i];     p.b1 = (const float*)d_in[i + 1];
    p.w2 = (const float*)d_in[i + 2]; p.b2 = (const float*)d_in[i + 3];
    p.w3 = (const float*)d_in[i + 4]; p.b3 = (const float*)d_in[i + 5];
    return p;
  };
  PP g1 = getPP(3), g2 = getPP(9), g3 = getPP(15), g4 = getPP(21), g5 = getPP(27);
  const float* k0_1  = (const float*)d_in[33];
  const float* amp_1 = (const float*)d_in[34];
  const float* k0_2  = (const float*)d_in[35];
  const float* amp_2 = (const float*)d_in[36];
  const float* d1w = (const float*)d_in[37];
  const float* d1b = (const float*)d_in[38];
  const float* d2w = (const float*)d_in[39];
  const float* d2b = (const float*)d_in[40];

  // workspace layout
  float* ws = (float*)d_ws;
  _Float16* Ftab = (_Float16*)d_ws;       // 3 * 16384 f16 = 98304 B = 24576 floats
  float* U    = ws + 24576;               // 2,097,152 (u 64ch; later kern/c)
  float* kern = U;                        // reuse (1,048,576)
  float* cbuf = U + 1048576;              // reuse (1,048,576)
  float* XL   = U + 2097152;              // 1,048,576
  float* A    = XL + 1048576;             // 2,097,152
  float* Bb   = A + 2097152;              // 2,097,152
  float* Cc   = Bb + 2097152;             // 2,097,152
  float* Dd   = Cc + 2097152;             // 2,097,152
  size_t need = (size_t)(Dd + 2097152 - ws) * sizeof(float);
  if (ws_size < need) return;

  const float invN2 = 1.0f / 16384.0f;

  twiddle_k<<<64, 256, 0, stream>>>(Ftab);
  stage_a<<<256, 256, 0, stream>>>(x, sos, den, g2, g4, g3, U, XL);

  // ---- Green's 1 (64 -> 32 channels) ----
  dft_pass<<<256, 256, 0, stream>>>(Ftab, U,  nullptr, A,  Bb, 64, 0, 0, 1.0f);
  dft_pass<<<256, 256, 0, stream>>>(Ftab, A,  Bb,      Cc, Dd, 64, 1, 1, 1.0f);
  freq_apply<<<128, 256, 0, stream>>>(Cc, Dd, A, Bb, k0_1, amp_1, 64);
  dft_pass<<<256, 256, 0, stream>>>(Ftab, A,  Bb,      Cc, Dd, 32, 2, 1, 1.0f);
  dft_pass<<<256, 256, 0, stream>>>(Ftab, Cc, Dd,      kern, nullptr, 32, 3, 0, invN2);

  // ---- Green's 2 (32 -> 32 channels) ----
  dft_pass<<<256, 256, 0, stream>>>(Ftab, kern, nullptr, A,  Bb, 32, 0, 0, 1.0f);
  dft_pass<<<256, 256, 0, stream>>>(Ftab, A,  Bb,        Cc, Dd, 32, 1, 1, 1.0f);
  freq_apply<<<128, 256, 0, stream>>>(Cc, Dd, A, Bb, k0_2, amp_2, 32);
  dft_pass<<<256, 256, 0, stream>>>(Ftab, A,  Bb,        Cc, Dd, 32, 2, 1, 1.0f);
  dft_pass<<<256, 256, 0, stream>>>(Ftab, Cc, Dd,        cbuf, nullptr, 32, 3, 0, invN2);

  stage_d<<<256, 256, 0, stream>>>(sos, den, cbuf, XL, g1, g5,
                                   d1w, d1b, d2w, d2b, (float*)d_out);
}
